// MoonViTAttention_58016418234517
// MI455X (gfx1250) — compile-verified, matrix operand order checked by reading, not run on hardware
//
#include <hip/hip_runtime.h>
#include <hip/hip_bf16.h>

typedef __bf16 bf16_t;
typedef __attribute__((ext_vector_type(16))) __bf16 v16bf;
typedef __attribute__((ext_vector_type(4)))  __bf16 v4bf;
typedef __attribute__((ext_vector_type(8)))  float  v8f;
typedef __attribute__((ext_vector_type(4)))  float  v4f_;

#define B_  8
#define S_  1024
#define H_  1024
#define NH_ 16
#define HD_ 64

// ---------------------------------------------------------------------------
// fp32 -> bf16 conversion, 4 elements / thread
// ---------------------------------------------------------------------------
__global__ void __launch_bounds__(256)
cvt_f32_bf16(const float* __restrict__ src, bf16_t* __restrict__ dst, int n4) {
    int i = blockIdx.x * blockDim.x + threadIdx.x;
    if (i < n4) {
        v4f_ f = *(const v4f_*)(src + (size_t)i * 4);
        *(v4bf*)(dst + (size_t)i * 4) = __builtin_convertvector(f, v4bf);
    }
}

// ---------------------------------------------------------------------------
// GEMM: C[M=8192, N=1024] = X(bf16) @ W(bf16)^T + bias
// Block: 256 threads = 8 waves arranged 2x4; block tile 128x128.
// Wave tile 64x32 = 4 M-subtiles x 2 N-subtiles -> 8 WMMAs per K step
// fed by 6 fragment loads (4 A + 2 B).
// A fragment: lane l holds row m=l%16, 16 contiguous k (half by l/16)
// B fragment: lane l holds col n=l%16 (row n of W), same k split
// C layout (ISA): lanes 0-15: N=lane, M=r; lanes 16-31: N=lane-16, M=8+r
// MODE 0: fp32 out, row-major [M,H]           (final projection)
// MODE 1: bf16 out, [B,NH,S,HD]               (Q, K before RoPE)
// MODE 2: bf16 out, [B,NH,HD,S] (transposed)  (V)
// ---------------------------------------------------------------------------
template <int MODE>
__global__ void __launch_bounds__(256)
gemm_bf16(const bf16_t* __restrict__ X, const bf16_t* __restrict__ W,
          const float* __restrict__ bias, void* __restrict__ out)
{
    const int wave = threadIdx.x >> 5;
    const int lane = threadIdx.x & 31;
    const int lh   = lane & 15;
    const int hi   = lane >> 4;
    const int NBLK = H_ / 128;               // 8 blocks along N
    const int bm   = blockIdx.x / NBLK;      // 0..63
    const int bn   = blockIdx.x % NBLK;
    const int wm   = wave >> 2;              // 0..1
    const int wn   = wave & 3;               // 0..3
    const int m0   = bm * 128 + wm * 64;
    const int n0   = bn * 128 + wn * 32;
    const int khalf = hi * 16;

    const bf16_t* xr0 = X + (size_t)(m0 + lh) * H_ + khalf;
    const bf16_t* xr1 = xr0 + (size_t)16 * H_;
    const bf16_t* xr2 = xr0 + (size_t)32 * H_;
    const bf16_t* xr3 = xr0 + (size_t)48 * H_;
    const bf16_t* wr0 = W + (size_t)(n0 + lh) * H_ + khalf;
    const bf16_t* wr1 = wr0 + (size_t)16 * H_;

    v8f acc[4][2] = {};
    for (int kk = 0; kk < H_; kk += 32) {
        const v16bf b0 = *(const v16bf*)(wr0 + kk);
        const v16bf b1 = *(const v16bf*)(wr1 + kk);
        const v16bf a0 = *(const v16bf*)(xr0 + kk);
        const v16bf a1 = *(const v16bf*)(xr1 + kk);
        const v16bf a2 = *(const v16bf*)(xr2 + kk);
        const v16bf a3 = *(const v16bf*)(xr3 + kk);
        if (kk + 32 < H_) {
            __builtin_prefetch(xr0 + kk + 32, 0, 1);   // global_prefetch_b8
            __builtin_prefetch(wr0 + kk + 32, 0, 1);
        }
        acc[0][0] = __builtin_amdgcn_wmma_f32_16x16x32_bf16(false, a0, false, b0, (short)0, acc[0][0], false, false);
        acc[0][1] = __builtin_amdgcn_wmma_f32_16x16x32_bf16(false, a0, false, b1, (short)0, acc[0][1], false, false);
        acc[1][0] = __builtin_amdgcn_wmma_f32_16x16x32_bf16(false, a1, false, b0, (short)0, acc[1][0], false, false);
        acc[1][1] = __builtin_amdgcn_wmma_f32_16x16x32_bf16(false, a1, false, b1, (short)0, acc[1][1], false, false);
        acc[2][0] = __builtin_amdgcn_wmma_f32_16x16x32_bf16(false, a2, false, b0, (short)0, acc[2][0], false, false);
        acc[2][1] = __builtin_amdgcn_wmma_f32_16x16x32_bf16(false, a2, false, b1, (short)0, acc[2][1], false, false);
        acc[3][0] = __builtin_amdgcn_wmma_f32_16x16x32_bf16(false, a3, false, b0, (short)0, acc[3][0], false, false);
        acc[3][1] = __builtin_amdgcn_wmma_f32_16x16x32_bf16(false, a3, false, b1, (short)0, acc[3][1], false, false);
    }

#pragma unroll
    for (int nt = 0; nt < 2; ++nt) {
        const int n  = n0 + nt * 16 + lh;
        const float bn_ = bias[n];
#pragma unroll
        for (int sub = 0; sub < 4; ++sub) {
#pragma unroll
            for (int r = 0; r < 8; ++r) {
                const int m = m0 + sub * 16 + r + 8 * hi;
                const float v = acc[sub][nt][r] + bn_;
                if (MODE == 0) {
                    ((float*)out)[(size_t)m * H_ + n] = v;
                } else {
                    const int b = m >> 10;            // m / S_
                    const int s = m & (S_ - 1);
                    const int h = n >> 6;             // n / HD_
                    const int d = n & (HD_ - 1);
                    if (MODE == 1) {
                        ((bf16_t*)out)[(((size_t)(b * NH_ + h)) * S_ + s) * HD_ + d] = (bf16_t)v;
                    } else {
                        ((bf16_t*)out)[(((size_t)(b * NH_ + h)) * HD_ + d) * S_ + s] = (bf16_t)v;
                    }
                }
            }
        }
    }
}

// ---------------------------------------------------------------------------
// In-place RoPE on bf16 Q and K stored as [B,NH,S,HD].
// One thread per (b,h,s,d<32) pair -> owns both halves, no races.
// Q additionally pre-scaled by 1/sqrt(HD) (exact power of two in bf16),
// so attention scores need no extra scaling.
// ---------------------------------------------------------------------------
__global__ void __launch_bounds__(256)
rope_inplace(bf16_t* __restrict__ q, bf16_t* __restrict__ k,
             const float* __restrict__ cosp, const float* __restrict__ sinp)
{
    const int i  = blockIdx.x * blockDim.x + threadIdx.x;  // B*NH*S*32 threads
    const int d  = i & 31;
    const int s  = (i >> 5) & (S_ - 1);
    const int bh = i >> 15;
    const size_t base = ((size_t)bh * S_ + s) * HD_;
    const float scale = 0.125f;                            // 1/sqrt(64)

    const float c1 = cosp[s * HD_ + d],      s1 = sinp[s * HD_ + d];
    const float c2 = cosp[s * HD_ + d + 32], s2 = sinp[s * HD_ + d + 32];

    const float q1 = (float)q[base + d], q2 = (float)q[base + d + 32];
    q[base + d]      = (bf16_t)((q1 * c1 - q2 * s1) * scale);
    q[base + d + 32] = (bf16_t)((q2 * c2 + q1 * s2) * scale);

    const float k1 = (float)k[base + d], k2 = (float)k[base + d + 32];
    k[base + d]      = (bf16_t)(k1 * c1 - k2 * s1);
    k[base + d + 32] = (bf16_t)(k2 * c2 + k1 * s2);
}

// ---------------------------------------------------------------------------
// Flash attention (online softmax), bf16 WMMA with f32 accumulate.
// Grid: B*NH * (S/128) blocks, 256 threads (8 waves). Wave w owns 16 q rows.
// Q,K: [B,NH,S,HD];  VT: [B,NH,HD,S];  Out: [B,S,NH,HD] (= [8192, H] rows).
//
// Per 32-key step:
//   - 4 score WMMAs (Q pre-scaled by 1/sqrt(HD))
//   - C-layout scores dumped to LDS, read back row-major (ds_load_b128 x4):
//     lane l then owns row l%16, keys (l/16)*16..+15 -- exactly the A-fragment
//     layout needed for the PV WMMA, so exp+pack happen in-register.
//   - full-row max/sum: single __shfl_xor(16) each (row lives in 2 lanes)
//   - 8 broadcasts rescale the C-layout O accumulators
//   - 4 PV WMMAs
// ---------------------------------------------------------------------------
__global__ void __launch_bounds__(256)
flash_attn(const bf16_t* __restrict__ Q, const bf16_t* __restrict__ K,
           const bf16_t* __restrict__ VT, bf16_t* __restrict__ Out)
{
    __shared__ float lds_s[8][16][32];     // per-wave 16x32 score tile, 16 KB

    const int wave = threadIdx.x >> 5;
    const int lane = threadIdx.x & 31;
    const int lh   = lane & 15;
    const int hi   = lane >> 4;
    const int bh   = blockIdx.x >> 3;      // b*NH + h
    const int qblk = blockIdx.x & 7;
    const int q0   = qblk * 128 + wave * 16;

    const bf16_t* qbase = Q  + ((size_t)bh * S_ + q0) * HD_;
    const bf16_t* kbase = K  + (size_t)bh * S_ * HD_;
    const bf16_t* vbase = VT + (size_t)bh * HD_ * S_;

    // Q tile as two 16x32 A fragments (head dim split 0..31 / 32..63)
    const v16bf qa0 = *(const v16bf*)(qbase + (size_t)lh * HD_ + hi * 16);
    const v16bf qa1 = *(const v16bf*)(qbase + (size_t)lh * HD_ + 32 + hi * 16);

    v8f o0 = {}, o1 = {}, o2 = {}, o3 = {};
    float mrun = -1e30f;                   // running max of row lh
    float lrun = 0.0f;                     // running sum of row lh

    for (int j = 0; j < S_; j += 32) {
        // K fragments: col n = key (j+lh / j+16+lh), k = head-dim slice
        const bf16_t* k0p = kbase + (size_t)(j + lh) * HD_ + hi * 16;
        const bf16_t* k1p = kbase + (size_t)(j + 16 + lh) * HD_ + hi * 16;
        const v16bf kb00 = *(const v16bf*)(k0p);
        const v16bf kb01 = *(const v16bf*)(k0p + 32);
        const v16bf kb10 = *(const v16bf*)(k1p);
        const v16bf kb11 = *(const v16bf*)(k1p + 32);

        v8f s0 = {}, s1 = {};
        s0 = __builtin_amdgcn_wmma_f32_16x16x32_bf16(false, qa0, false, kb00, (short)0, s0, false, false);
        s0 = __builtin_amdgcn_wmma_f32_16x16x32_bf16(false, qa1, false, kb01, (short)0, s0, false, false);
        s1 = __builtin_amdgcn_wmma_f32_16x16x32_bf16(false, qa0, false, kb10, (short)0, s1, false, false);
        s1 = __builtin_amdgcn_wmma_f32_16x16x32_bf16(false, qa1, false, kb11, (short)0, s1, false, false);

        // C-layout -> LDS (row M = r+8*hi, key col lh / lh+16)
#pragma unroll
        for (int r = 0; r < 8; ++r) {
            lds_s[wave][r + 8 * hi][lh]      = s0[r];
            lds_s[wave][r + 8 * hi][lh + 16] = s1[r];
        }

        // read back row-major: lane owns row lh, keys hi*16..hi*16+15
        float vals[16];
        const float* rowp = &lds_s[wave][lh][hi * 16];
#pragma unroll
        for (int t = 0; t < 4; ++t) {
            const v4f_ v = *(const v4f_*)(rowp + 4 * t);
            vals[4 * t + 0] = v.x; vals[4 * t + 1] = v.y;
            vals[4 * t + 2] = v.z; vals[4 * t + 3] = v.w;
        }

        float mx = vals[0];
#pragma unroll
        for (int i = 1; i < 16; ++i) mx = fmaxf(mx, vals[i]);
        mx = fmaxf(mx, __shfl_xor(mx, 16, 32));        // other half of row

        const float mnew = fmaxf(mrun, mx);
        const float corr = __expf(mrun - mnew);
        mrun = mnew;

        v16bf pa = {};
        float ps = 0.0f;
#pragma unroll
        for (int i = 0; i < 16; ++i) {
            const float p = __expf(vals[i] - mnew);
            ps += p;
            pa[i] = (bf16_t)p;
        }
        ps += __shfl_xor(ps, 16, 32);
        lrun = lrun * corr + ps;

        // rescale O accumulators (C layout: component r <-> row r+8*hi)
#pragma unroll
        for (int r = 0; r < 8; ++r) {
            const float cb = __shfl(corr, r + 8 * hi, 32);
            o0[r] *= cb; o1[r] *= cb; o2[r] *= cb; o3[r] *= cb;
        }

        // O += P(16x32) @ V(32x16) for four head-dim column tiles.
        // B fragment from VT: lane holds dim row (n), 16 contiguous keys.
        v16bf vb;
        vb = *(const v16bf*)(vbase + (size_t)(lh)      * S_ + j + hi * 16);
        o0 = __builtin_amdgcn_wmma_f32_16x16x32_bf16(false, pa, false, vb, (short)0, o0, false, false);
        vb = *(const v16bf*)(vbase + (size_t)(16 + lh) * S_ + j + hi * 16);
        o1 = __builtin_amdgcn_wmma_f32_16x16x32_bf16(false, pa, false, vb, (short)0, o1, false, false);
        vb = *(const v16bf*)(vbase + (size_t)(32 + lh) * S_ + j + hi * 16);
        o2 = __builtin_amdgcn_wmma_f32_16x16x32_bf16(false, pa, false, vb, (short)0, o2, false, false);
        vb = *(const v16bf*)(vbase + (size_t)(48 + lh) * S_ + j + hi * 16);
        o3 = __builtin_amdgcn_wmma_f32_16x16x32_bf16(false, pa, false, vb, (short)0, o3, false, false);
    }

    // normalize and write [B,S,NH,HD]
    const float linv = 1.0f / lrun;        // per lane: row lh
    const int b = bh >> 4;
    const int h = bh & 15;
#pragma unroll
    for (int r = 0; r < 8; ++r) {
        const float li = __shfl(linv, r + 8 * hi, 32);
        const int qrow = q0 + r + 8 * hi;
        const size_t rowbase = (((size_t)b * S_ + qrow) * NH_ + h) * (size_t)HD_;
        Out[rowbase + lh]      = (bf16_t)(o0[r] * li);
        Out[rowbase + 16 + lh] = (bf16_t)(o1[r] * li);
        Out[rowbase + 32 + lh] = (bf16_t)(o2[r] * li);
        Out[rowbase + 48 + lh] = (bf16_t)(o3[r] * li);
    }
}

// ---------------------------------------------------------------------------
extern "C" void kernel_launch(void* const* d_in, const int* in_sizes, int n_in,
                              void* d_out, int out_size, void* d_ws, size_t ws_size,
                              hipStream_t stream)
{
    (void)in_sizes; (void)n_in; (void)out_size; (void)ws_size;

    const float* hs   = (const float*)d_in[0];
    const float* cosp = (const float*)d_in[1];
    const float* sinp = (const float*)d_in[2];
    const float* Wq   = (const float*)d_in[3];
    const float* bq   = (const float*)d_in[4];
    const float* Wk   = (const float*)d_in[5];
    const float* bk   = (const float*)d_in[6];
    const float* Wv   = (const float*)d_in[7];
    const float* bv   = (const float*)d_in[8];
    const float* Wo   = (const float*)d_in[9];
    const float* bo   = (const float*)d_in[10];

    const size_t MN = (size_t)B_ * S_ * H_;   // 8388608
    const size_t HH = (size_t)H_ * H_;        // 1048576

    char* ws = (char*)d_ws;
    size_t off = 0;
    bf16_t* hb  = (bf16_t*)(ws + off); off += MN * sizeof(bf16_t);
    bf16_t* wqb = (bf16_t*)(ws + off); off += HH * sizeof(bf16_t);
    bf16_t* wkb = (bf16_t*)(ws + off); off += HH * sizeof(bf16_t);
    bf16_t* wvb = (bf16_t*)(ws + off); off += HH * sizeof(bf16_t);
    bf16_t* wob = (bf16_t*)(ws + off); off += HH * sizeof(bf16_t);
    bf16_t* qb  = (bf16_t*)(ws + off); off += MN * sizeof(bf16_t);
    bf16_t* kb  = (bf16_t*)(ws + off); off += MN * sizeof(bf16_t);
    bf16_t* vt  = (bf16_t*)(ws + off); off += MN * sizeof(bf16_t);
    bf16_t* ab  = (bf16_t*)(ws + off); off += MN * sizeof(bf16_t);

    // 1) convert activations + weights to bf16 (4 elems / thread)
    cvt_f32_bf16<<<(int)(MN / 4 / 256), 256, 0, stream>>>(hs, hb, (int)(MN / 4));
    cvt_f32_bf16<<<(int)(HH / 4 / 256), 256, 0, stream>>>(Wq, wqb, (int)(HH / 4));
    cvt_f32_bf16<<<(int)(HH / 4 / 256), 256, 0, stream>>>(Wk, wkb, (int)(HH / 4));
    cvt_f32_bf16<<<(int)(HH / 4 / 256), 256, 0, stream>>>(Wv, wvb, (int)(HH / 4));
    cvt_f32_bf16<<<(int)(HH / 4 / 256), 256, 0, stream>>>(Wo, wob, (int)(HH / 4));

    // 2) projections: (8192/128)*(1024/128) = 512 blocks of 8 waves
    const int gemm_blocks = (B_ * S_ / 128) * (H_ / 128);
    gemm_bf16<1><<<gemm_blocks, 256, 0, stream>>>(hb, wqb, bq, (void*)qb);
    gemm_bf16<1><<<gemm_blocks, 256, 0, stream>>>(hb, wkb, bk, (void*)kb);
    gemm_bf16<2><<<gemm_blocks, 256, 0, stream>>>(hb, wvb, bv, (void*)vt);

    // 3) RoPE on Q,K (Q pre-scaled by 1/sqrt(HD)): B*NH*S*32 threads
    const int rope_threads = B_ * NH_ * S_ * 32;
    rope_inplace<<<rope_threads / 256, 256, 0, stream>>>(qb, kb, cosp, sinp);

    // 4) attention: B*NH*(S/128) blocks
    flash_attn<<<B_ * NH_ * (S_ / 128), 256, 0, stream>>>(qb, kb, vt, ab);

    // 5) output projection -> fp32 d_out
    gemm_bf16<0><<<gemm_blocks, 256, 0, stream>>>(ab, wob, bo, d_out);
}
